// ResidualBlock_7662221656703
// MI455X (gfx1250) — compile-verified, matrix operand order checked by reading, not run on hardware
//
#include <hip/hip_runtime.h>
#include <stdint.h>

// ---------------- types ----------------
typedef __bf16 v16bf __attribute__((ext_vector_type(16)));
typedef float  v8f   __attribute__((ext_vector_type(8)));

#define BM 128
#define BN 256
#define BK 32
#define LDS_STRIDE 40   // bf16 elements per LDS row (32 data + 8 pad -> 80B stride, conflict-free)

#define D_DIM 1024
#define T_DIM 4096
#define B_DIM 4
#define NROWS (B_DIM * T_DIM)          // 16384
#define NELEM (NROWS * D_DIM)          // 16777216
#define SEGS 32
#define SEGLEN (T_DIM / SEGS)          // 128

__device__ __forceinline__ unsigned short f2bf(float f) {
    unsigned u = __float_as_uint(f);
    u += 0x7fffu + ((u >> 16) & 1u);   // round-to-nearest-even
    return (unsigned short)(u >> 16);
}

__device__ __forceinline__ float gelu_exact(float x) {
    return 0.5f * x * (1.0f + erff(x * 0.70710678118654752f));
}
__device__ __forceinline__ float sigmoidf_(float x) {
    return 1.0f / (1.0f + expf(-x));
}

// ---- CDNA5 async global->LDS (ASYNCcnt) ----
typedef __attribute__((address_space(3))) const unsigned short* lds_cptr_u16;

__device__ __forceinline__ unsigned lds_off(const unsigned short* p) {
    // generic (flat) shared pointer -> LDS byte address
    return (unsigned)(unsigned long long)(lds_cptr_u16)p;
}

__device__ __forceinline__ void async_b128(unsigned lds_byte_addr, const unsigned short* g) {
    asm volatile("global_load_async_to_lds_b128 %0, %1, off"
                 :: "v"(lds_byte_addr), "v"(g) : "memory");
}

__device__ __forceinline__ void wait_async0() {
#if defined(__has_builtin)
#if __has_builtin(__builtin_amdgcn_s_wait_asynccnt)
    __builtin_amdgcn_s_wait_asynccnt(0);
    return;
#endif
#endif
    asm volatile("s_wait_asynccnt 0x0" ::: "memory");
}

// ---------------- block row-reduction (sum) ----------------
__device__ __forceinline__ float block_reduce_sum(float v) {
    for (int o = 16; o > 0; o >>= 1) v += __shfl_xor(v, o);
    __shared__ float red[8];
    __shared__ float total;
    if ((threadIdx.x & 31) == 0) red[threadIdx.x >> 5] = v;
    __syncthreads();
    if (threadIdx.x == 0) {
        float t = 0.f;
        for (int i = 0; i < 8; ++i) t += red[i];
        total = t;
    }
    __syncthreads();
    return total;
}

// ================= GEMM: C[M,N] = A[M,K] @ W[N,K]^T + bias =================
// A, W bf16 (raw u16 bits), C f32.  v_wmma_f32_16x16x32_bf16, 64x64 per wave,
// 128x256 per block, double-buffered async global->LDS tile loads.
__global__ __launch_bounds__(256) void k_gemm_bf16(
    const unsigned short* __restrict__ A,
    const unsigned short* __restrict__ W,
    const float* __restrict__ bias,
    float* __restrict__ C,
    int M, int N, int K)
{
    __shared__ __attribute__((aligned(16))) unsigned short sA[2][BM * LDS_STRIDE];
    __shared__ __attribute__((aligned(16))) unsigned short sB[2][BN * LDS_STRIDE];

    const int tid  = threadIdx.x;
    const int lane = tid & 31;
    const int wave = tid >> 5;     // 0..7
    const int wm   = wave >> 2;    // 0..1 (M)
    const int wn   = wave & 3;     // 0..3 (N)
    const int l16  = lane & 15;
    const int lhalf= lane >> 4;

    const int m0 = blockIdx.y * BM;
    const int n0 = blockIdx.x * BN;

    // A tile: 128 rows x 64B = 8KB -> 2 threads/row, 32B each
    const int arow   = tid >> 1;
    const int achunk = tid & 1;
    // B tile: 256 rows x 64B = 16KB -> 1 thread/row, 64B each

    v8f acc[4][4];
    const v8f vzero = {0.f,0.f,0.f,0.f,0.f,0.f,0.f,0.f};
#pragma unroll
    for (int i = 0; i < 4; ++i)
#pragma unroll
        for (int j = 0; j < 4; ++j) acc[i][j] = vzero;

    union Frag { uint4 q[2]; v16bf v; };

    // issue async loads for the tile at k-offset kt into buffer `buf`
    auto issue_tile = [&](int kt, int buf) {
        const unsigned short* gA = A + (size_t)(m0 + arow) * K + kt + achunk * 16;
        unsigned la = lds_off(&sA[buf][arow * LDS_STRIDE + achunk * 16]);
        async_b128(la,      gA);
        async_b128(la + 16, gA + 8);
        const unsigned short* gB = W + (size_t)(n0 + tid) * K + kt;
        unsigned lb = lds_off(&sB[buf][tid * LDS_STRIDE]);
        async_b128(lb,      gB);
        async_b128(lb + 16, gB + 8);
        async_b128(lb + 32, gB + 16);
        async_b128(lb + 48, gB + 24);
    };

    issue_tile(0, 0);
    int buf = 0;

    for (int kt = 0; kt < K; kt += BK) {
        wait_async0();          // my async loads for tile kt complete
        __syncthreads();        // everyone's loads done; everyone done with prior compute
        if (kt + BK < K) issue_tile(kt + BK, buf ^ 1);

        const unsigned short* cA = &sA[buf][0];
        const unsigned short* cB = &sB[buf][0];

        Frag fb[4];
#pragma unroll
        for (int j = 0; j < 4; ++j) {
            const unsigned short* p = cB + (wn * 64 + j * 16 + l16) * LDS_STRIDE;
            fb[j].q[0] = *(const uint4*)(p + lhalf * 8);        // K 0..7 / 8..15
            fb[j].q[1] = *(const uint4*)(p + 16 + lhalf * 8);   // K 16..23 / 24..31
        }
#pragma unroll
        for (int i = 0; i < 4; ++i) {
            Frag fa;
            const unsigned short* p = cA + (wm * 64 + i * 16 + l16) * LDS_STRIDE;
            fa.q[0] = *(const uint4*)(p + lhalf * 8);
            fa.q[1] = *(const uint4*)(p + 16 + lhalf * 8);
#pragma unroll
            for (int j = 0; j < 4; ++j)
                acc[i][j] = __builtin_amdgcn_wmma_f32_16x16x32_bf16(
                    false, fa.v, false, fb[j].v, (short)0, acc[i][j], false, false);
        }
        buf ^= 1;
    }

    // epilogue: C/D layout — VGPR r: lanes 0-15 M=r, lanes 16-31 M=8+r; N=lane&15
#pragma unroll
    for (int j = 0; j < 4; ++j) {
        const int col = n0 + wn * 64 + j * 16 + l16;
        const float bc = bias ? bias[col] : 0.f;
#pragma unroll
        for (int i = 0; i < 4; ++i) {
            const int rbase = m0 + wm * 64 + i * 16 + lhalf * 8;
            float* cp = C + (size_t)rbase * N + col;
#pragma unroll
            for (int r = 0; r < 8; ++r)
                cp[(size_t)r * N] = acc[i][j][r] + bc;
        }
    }
}

// ================= elementwise / norm / scan kernels =================
__global__ void k_f32_to_bf16(const float* __restrict__ src,
                              unsigned short* __restrict__ dst, int n) {
    int i = blockIdx.x * 256 + threadIdx.x;
    if (i < n) dst[i] = f2bf(src[i]);
}

// rmsnorm of x with gain g -> bf16
__global__ __launch_bounds__(256) void k_rmsnorm_bf16(
    const float* __restrict__ x, const float* __restrict__ g,
    unsigned short* __restrict__ out) {
    const size_t off = (size_t)blockIdx.x * D_DIM;
    float v[4]; float ss = 0.f;
#pragma unroll
    for (int i = 0; i < 4; ++i) {
        v[i] = x[off + threadIdx.x + i * 256];
        ss += v[i] * v[i];
    }
    float tot = block_reduce_sum(ss);
    float scale = 32.0f / fmaxf(sqrtf(tot), 1e-12f);   // sqrt(1024)=32
#pragma unroll
    for (int i = 0; i < 4; ++i) {
        int d = threadIdx.x + i * 256;
        out[off + d] = f2bf(v[i] * scale * g[d]);
    }
}

// residual = xr + x (store f32), then rmsnorm(residual, g) -> bf16
__global__ __launch_bounds__(256) void k_addres_rmsnorm(
    const float* __restrict__ xr, const float* __restrict__ x,
    const float* __restrict__ g, float* __restrict__ res_out,
    unsigned short* __restrict__ nrm_out) {
    const size_t off = (size_t)blockIdx.x * D_DIM;
    float v[4]; float ss = 0.f;
#pragma unroll
    for (int i = 0; i < 4; ++i) {
        int d = threadIdx.x + i * 256;
        v[i] = xr[off + d] + x[off + d];
        res_out[off + d] = v[i];
        ss += v[i] * v[i];
    }
    float tot = block_reduce_sum(ss);
    float scale = 32.0f / fmaxf(sqrtf(tot), 1e-12f);
#pragma unroll
    for (int i = 0; i < 4; ++i) {
        int d = threadIdx.x + i * 256;
        nrm_out[off + d] = f2bf(v[i] * scale * g[d]);
    }
}

// gel = gelu(lin) f32; also bf16(lin) for next GEMM
__global__ void k_gelu_cvt(const float* __restrict__ lin,
                           float* __restrict__ gel,
                           unsigned short* __restrict__ lin_bf, int n) {
    int i = blockIdx.x * 256 + threadIdx.x;
    if (i >= n) return;
    float v = lin[i];
    gel[i] = gelu_exact(v);
    lin_bf[i] = f2bf(v);
}

// RG-LRU gates: a = exp(log_a * sigmoid(rt)/8), gated = sqrt(1-a^2)*sigmoid(it)*x
__global__ void k_gates(const float* __restrict__ rt_pre,
                        const float* __restrict__ it_pre,
                        const float* __restrict__ xrg,
                        const float* __restrict__ Lam,
                        float* __restrict__ a_out, float* __restrict__ g_out, int n) {
    int i = blockIdx.x * 256 + threadIdx.x;
    if (i >= n) return;
    int d = i & (D_DIM - 1);
    float rt = sigmoidf_(rt_pre[i]);
    float it = sigmoidf_(it_pre[i]);
    float la = -log1pf(expf(-Lam[d]));      // -softplus(-Lam)
    float a  = expf(la * rt * 0.125f);
    float gt = sqrtf(fmaxf(1.0f - a * a, 0.f)) * (it * xrg[i]);
    a_out[i] = a;
    g_out[i] = gt;
}

// --- segmented linear scan h_t = a_t h_{t-1} + g_t ---
// pass1: per (b, seg, d): local scan from h_in=0; store h_local + cumulative product c
__global__ void k_scan_pass1(const float* __restrict__ a, const float* __restrict__ g,
                             float* __restrict__ hl, float* __restrict__ c) {
    int p = blockIdx.x * 256 + threadIdx.x;           // [0, B*SEGS*D)
    int d = p & (D_DIM - 1);
    int s = (p >> 10) & (SEGS - 1);
    int b = p >> 15;
    size_t base = ((size_t)b * T_DIM + (size_t)s * SEGLEN) * D_DIM + d;
    float h = 0.f, cp = 1.f;
    for (int t = 0; t < SEGLEN; ++t) {
        size_t idx = base + (size_t)t * D_DIM;
        float av = a[idx];
        cp *= av;
        h = av * h + g[idx];
        hl[idx] = h;
        c[idx]  = cp;
    }
}

// pass2: per (b,d) scan the SEGS carries: carry[s] = h at segment entry
__global__ void k_scan_pass2(const float* __restrict__ hl, const float* __restrict__ c,
                             float* __restrict__ carry) {
    int p = blockIdx.x * 256 + threadIdx.x;           // [0, B*D)
    int d = p & (D_DIM - 1);
    int b = p >> 10;
    float h_in = 0.f;
    for (int s = 0; s < SEGS; ++s) {
        carry[((size_t)b * SEGS + s) * D_DIM + d] = h_in;
        size_t idx_end = ((size_t)b * T_DIM + (size_t)s * SEGLEN + (SEGLEN - 1)) * D_DIM + d;
        h_in = c[idx_end] * h_in + hl[idx_end];
    }
}

// pass3 fused with product epilogue: prod_bf16 = bf16(gel * (h_local + c*carry))
__global__ void k_scan_fix_mul_cvt(const float* __restrict__ hl, const float* __restrict__ c,
                                   const float* __restrict__ carry,
                                   const float* __restrict__ gel,
                                   unsigned short* __restrict__ out_bf, int n) {
    int i = blockIdx.x * 256 + threadIdx.x;
    if (i >= n) return;
    int d = i & (D_DIM - 1);
    int t = (i >> 10) & (T_DIM - 1);
    int b = i >> 22;
    int s = t / SEGLEN;
    float cr = carry[((size_t)b * SEGS + s) * D_DIM + d];
    float h  = hl[i] + c[i] * cr;
    out_bf[i] = f2bf(gel[i] * h);
}

// combined = gelu(sigmoid(o1)) * o2 -> bf16
__global__ void k_sig_gelu_mul_cvt(const float* __restrict__ o1,
                                   const float* __restrict__ o2,
                                   unsigned short* __restrict__ out_bf, int n) {
    int i = blockIdx.x * 256 + threadIdx.x;
    if (i >= n) return;
    float s = sigmoidf_(o1[i]);
    out_bf[i] = f2bf(gelu_exact(s) * o2[i]);
}

__global__ void k_add(const float* __restrict__ y, const float* __restrict__ r,
                      float* __restrict__ out, int n) {
    int i = blockIdx.x * 256 + threadIdx.x;
    if (i < n) out[i] = y[i] + r[i];
}

// ================= launch =================
extern "C" void kernel_launch(void* const* d_in, const int* in_sizes, int n_in,
                              void* d_out, int out_size, void* d_ws, size_t ws_size,
                              hipStream_t stream) {
    (void)in_sizes; (void)n_in; (void)out_size; (void)ws_size;
    const float* x      = (const float*)d_in[0];
    const float* g1     = (const float*)d_in[1];
    const float* g2     = (const float*)d_in[2];
    const float* W_lin  = (const float*)d_in[3];
    const float* b_lin  = (const float*)d_in[4];
    const float* W_conv = (const float*)d_in[5];
    const float* b_conv = (const float*)d_in[6];
    const float* W_lin2 = (const float*)d_in[7];
    const float* b_lin2 = (const float*)d_in[8];
    const float* Wa     = (const float*)d_in[9];
    const float* ba     = (const float*)d_in[10];
    const float* Wx     = (const float*)d_in[11];
    const float* bx     = (const float*)d_in[12];
    const float* Lam    = (const float*)d_in[13];
    const float* W1     = (const float*)d_in[14];
    const float* b1     = (const float*)d_in[15];
    const float* W2     = (const float*)d_in[16];
    const float* b2     = (const float*)d_in[17];
    const float* W3     = (const float*)d_in[18];
    const float* b3     = (const float*)d_in[19];
    float* out = (float*)d_out;

    char* ws = (char*)d_ws;
    const size_t MB = 1024ull * 1024ull;
    // weights (bf16): 8 x 2MB
    const float* wsrc[8] = {W_lin, W_conv, W_lin2, Wa, Wx, W1, W2, W3};
    unsigned short* wb[8];
    for (int i = 0; i < 8; ++i) wb[i] = (unsigned short*)(ws + (size_t)i * 2 * MB);
    unsigned short* ABF = (unsigned short*)(ws + 16 * MB);   // 32MB bf16 activations
    float* F1 = (float*)(ws + 48  * MB);
    float* F2 = (float*)(ws + 112 * MB);
    float* F3 = (float*)(ws + 176 * MB);
    float* F4 = (float*)(ws + 240 * MB);
    float* CARRY = (float*)(ws + 304 * MB);                  // B*SEGS*D f32 = 512KB
    float* F5 = (float*)(ws + 305 * MB);                     // 64MB

    const int NE = NELEM;
    const int EB = (NE + 255) / 256;
    dim3 gG(D_DIM / BN, NROWS / BM);     // (4, 128)

    // weights -> bf16
    for (int i = 0; i < 8; ++i)
        k_f32_to_bf16<<<(D_DIM * D_DIM + 255) / 256, 256, 0, stream>>>(wsrc[i], wb[i], D_DIM * D_DIM);

    // xn = rmsnorm(x, g1) -> bf16
    k_rmsnorm_bf16<<<NROWS, 256, 0, stream>>>(x, g1, ABF);
    // lin = xn @ W_lin^T + b_lin
    k_gemm_bf16<<<gG, 256, 0, stream>>>(ABF, wb[0], b_lin, F1, NROWS, D_DIM, D_DIM);
    // gel = gelu(lin); lin -> bf16
    k_gelu_cvt<<<EB, 256, 0, stream>>>(F1, F2, ABF, NE);
    // xrg = lin @ W_conv^T + b_conv
    k_gemm_bf16<<<gG, 256, 0, stream>>>(ABF, wb[1], b_conv, F3, NROWS, D_DIM, D_DIM);
    k_f32_to_bf16<<<EB, 256, 0, stream>>>(F3, ABF, NE);
    // gate pre-activations
    k_gemm_bf16<<<gG, 256, 0, stream>>>(ABF, wb[3], ba, F1, NROWS, D_DIM, D_DIM);   // rt_pre
    k_gemm_bf16<<<gG, 256, 0, stream>>>(ABF, wb[4], bx, F4, NROWS, D_DIM, D_DIM);   // it_pre
    // a -> F1, gated -> F4 (in place)
    k_gates<<<EB, 256, 0, stream>>>(F1, F4, F3, Lam, F1, F4, NE);
    // segmented scan: h_local -> F3, cumprod -> F5, carries -> CARRY
    k_scan_pass1<<<(B_DIM * SEGS * D_DIM) / 256, 256, 0, stream>>>(F1, F4, F3, F5);
    k_scan_pass2<<<(B_DIM * D_DIM) / 256, 256, 0, stream>>>(F3, F5, CARRY);
    // fixup + multiply by gel -> bf16
    k_scan_fix_mul_cvt<<<EB, 256, 0, stream>>>(F3, F5, CARRY, F2, ABF, NE);
    // xr = (gel * h) @ W_lin2^T + b_lin2
    k_gemm_bf16<<<gG, 256, 0, stream>>>(ABF, wb[2], b_lin2, F1, NROWS, D_DIM, D_DIM);
    // residual = xr + x -> F2; xm = rmsnorm(residual, g2) -> bf16
    k_addres_rmsnorm<<<NROWS, 256, 0, stream>>>(F1, x, g2, F2, ABF);
    // MLP
    k_gemm_bf16<<<gG, 256, 0, stream>>>(ABF, wb[5], b1, F3, NROWS, D_DIM, D_DIM);   // o1
    k_gemm_bf16<<<gG, 256, 0, stream>>>(ABF, wb[6], b2, F4, NROWS, D_DIM, D_DIM);   // o2
    k_sig_gelu_mul_cvt<<<EB, 256, 0, stream>>>(F3, F4, ABF, NE);
    k_gemm_bf16<<<gG, 256, 0, stream>>>(ABF, wb[7], b3, F1, NROWS, D_DIM, D_DIM);   // y
    // out = y + residual
    k_add<<<EB, 256, 0, stream>>>(F1, F2, out, NE);
}
